// LTVMinimumPhaseFIRFilterPrecise_22041772163075
// MI455X (gfx1250) — matrix operand order repaired and checked by reading, not compiled, not run on hardware
//
#include <hip/hip_runtime.h>

// LTV minimum-phase FIR filter (precise), CDNA5/gfx1250.
// Stage 1: per-(b,frame) minimum-phase kernel via LDS-table DFTs (VALU).
// Stage 2: time-varying FIR as WMMA GEMM with split-f16 error compensation
//          (3x v_wmma_f32_16x16x32_f16 per K-chunk, f32 accumulate).
//          Taps staged to LDS with GLOBAL_LOAD_ASYNC_TO_LDS_B128 (ASYNCcnt);
//          x window stored REVERSED in LDS so A-operand 8-half groups are
//          contiguous ascending (no v_mov_b16 reversal swaps).

#define BATCH   4
#define FRAMES  400
#define NBINS   129
#define HOP     240
#define TLEN    (FRAMES * HOP)   // 96000
#define KLEN    256              // FFT size == FIR length

#define GSEG    15               // frames advanced per group (16 cols => lo&hi inside)
#define NGRP    27               // ceil(400/15)
#define NSPLIT  8                // time-splits per group for parallelism
#define WINMAX  736              // ceil(233/8)=30 tiles -> 30*16+255=735

typedef __attribute__((ext_vector_type(16))) _Float16 v16h;
typedef __attribute__((ext_vector_type(8)))  float    v8f;
typedef __attribute__((ext_vector_type(4)))  int      v4i;

typedef __attribute__((address_space(1))) v4i as1_v4i;
typedef __attribute__((address_space(3))) v4i as3_v4i;

// ---------------------------------------------------------------------------
// Stage 1: minimum-phase kernel design. One block per (b, frame).
// ---------------------------------------------------------------------------
__global__ __launch_bounds__(256) void minphase_stage1(
    const float* __restrict__ log_mag,   // (B, FRAMES, NBINS)
    _Float16* __restrict__ ker_hi,       // (B*FRAMES, 256)
    _Float16* __restrict__ ker_lo)       // (B*FRAMES, 256)
{
  __shared__ float ctab[256];
  __shared__ float stab[256];
  __shared__ float full[256];
  __shared__ float Xh[132];
  __shared__ float mp[256];
  __shared__ float cre[256];
  __shared__ float cim[256];

  const int p   = blockIdx.x;            // b*FRAMES + f
  const int tid = threadIdx.x;
  const float* lm = log_mag + (size_t)p * NBINS;

  const float ang = 6.283185307179586f * (float)tid * (1.0f / 256.0f);
  ctab[tid] = cosf(ang);
  stab[tid] = sinf(ang);
  // full = [log_mag, flip(log_mag)[1:-1]] : full[n] = lm[256-n] for n>128
  full[tid] = (tid <= 128) ? lm[tid] : lm[256 - tid];
  __syncthreads();

  // Xf (real, even input) times Hilbert mask h = {1, 2x127, 1, 0x127}
  if (tid <= 128) {
    float acc = 0.0f;
    for (int n = 0; n < 256; ++n)
      acc += full[n] * ctab[(n * tid) & 255];
    const float hk = (tid == 0 || tid == 128) ? 1.0f : 2.0f;
    Xh[tid] = hk * acc;
  }
  __syncthreads();

  // min_phase[n] = -(1/256) * sum_{k=0..128} Xh[k] * sin(2pi n k / 256)
  {
    float acc = 0.0f;
    for (int k = 0; k <= 128; ++k)
      acc += Xh[k] * stab[(tid * k) & 255];
    mp[tid] = -acc * (1.0f / 256.0f);
  }
  __syncthreads();

  {
    const float e = expf(full[tid]);
    float sn, cn;
    sincosf(mp[tid], &sn, &cn);
    cre[tid] = e * cn;
    cim[tid] = e * sn;
  }
  __syncthreads();

  {
    float acc = 0.0f;
    for (int k = 0; k < 256; ++k) {
      const int m = (tid * k) & 255;
      acc += cre[k] * ctab[m] - cim[k] * stab[m];
    }
    float ker = acc * (1.0f / 256.0f);
    const float wnd = (tid < 128) ? 1.0f : (0.5f - 0.5f * ctab[tid]); // half-Hann
    ker *= wnd;
    const _Float16 h = (_Float16)ker;
    const _Float16 l = (_Float16)(ker - (float)h);
    ker_hi[(size_t)p * 256 + tid] = h;
    ker_lo[(size_t)p * 256 + tid] = l;
  }
}

// ---------------------------------------------------------------------------
// Stage 2: out[b,t] = (1-w)*y_lo[t] + w*y_hi[t],  y_f = x (*) kernel[f].
// WMMA tile: M=16 timesteps x N=16 frames (f0..f0+15), K accumulated 32/chunk.
// ---------------------------------------------------------------------------
__global__ __launch_bounds__(256) void ltv_fir_stage2(
    const float*    __restrict__ ex,      // (B, TLEN)
    const _Float16* __restrict__ ker_hi,  // (B*FRAMES, 256)
    const _Float16* __restrict__ ker_lo,
    float*          __restrict__ out)     // (B, TLEN)
{
  __shared__ _Float16 xrhi[WINMAX];                             // reversed x (hi)
  __shared__ _Float16 xrlo[WINMAX];                             // reversed x (lo)
  __shared__ _Float16 khi[16 * 256] __attribute__((aligned(32)));
  __shared__ _Float16 klo[16 * 256] __attribute__((aligned(32)));

  const int g  = blockIdx.x;              // frame group
  const int b  = blockIdx.y;              // batch
  const int z  = blockIdx.z;              // time split
  const int f0 = g * GSEG;

  // t-range with lo(t) in [15g, 15g+14]; group 0 also owns the clipped head.
  const int t_start = (g == 0) ? 0 : (HOP * GSEG * g + HOP / 2);
  const int t_end   = min(HOP * (GSEG * g + GSEG - 1) + (HOP + HOP / 2 - 1), TLEN - 1);
  const int count   = t_end - t_start + 1;
  const int ntiles  = (count + 15) >> 4;

  const int tiles_per = (ntiles + NSPLIT - 1) / NSPLIT;
  const int tile_lo   = z * tiles_per;
  const int tile_hi   = min(tile_lo + tiles_per, ntiles);
  const int ntl       = max(tile_hi - tile_lo, 0);

  const int t_sub   = t_start + (tile_lo << 4);
  const int win_lo  = t_sub - (KLEN - 1);
  const int win_len = ntl * 16 + (KLEN - 1);

  const int tid = threadIdx.x;

  // ---- Stage taps (already split-f16 in global) into LDS -------------------
#if __has_builtin(__builtin_amdgcn_global_load_async_to_lds_b128)
  #pragma unroll
  for (int it = 0; it < 2; ++it) {
    const int ho = (it * 256 + tid) * 8;          // half offset, 8 halves = 16B/lane
    const int fr = ho >> 8;
    const int f  = min(f0 + fr, FRAMES - 1);
    const size_t src = ((size_t)b * FRAMES + f) * 256 + (ho & 255);
    __builtin_amdgcn_global_load_async_to_lds_b128(
        (as1_v4i*)(uintptr_t)(ker_hi + src),
        (as3_v4i*)(uintptr_t)(khi + ho), 0, 0);
    __builtin_amdgcn_global_load_async_to_lds_b128(
        (as1_v4i*)(uintptr_t)(ker_lo + src),
        (as3_v4i*)(uintptr_t)(klo + ho), 0, 0);
  }
#else
  if (tid == 0) {
    __builtin_prefetch(ker_hi + ((size_t)b * FRAMES + f0) * 256, 0, 1);
    __builtin_prefetch(ker_lo + ((size_t)b * FRAMES + f0) * 256, 0, 1);
  }
  for (int i = tid; i < 16 * 256; i += 256) {
    const int fr = i >> 8;
    const int f  = min(f0 + fr, FRAMES - 1);
    const size_t src = ((size_t)b * FRAMES + f) * 256 + (i & 255);
    khi[i] = ker_hi[src];
    klo[i] = ker_lo[src];
  }
#endif

  // ---- Stage x window REVERSED into LDS as split f16 -----------------------
  for (int i = tid; i < win_len; i += 256) {
    const int t = win_lo + i;
    const float v = (t >= 0 && t < TLEN) ? ex[(size_t)b * TLEN + t] : 0.0f;
    const _Float16 h = (_Float16)v;
    const int r = (win_len - 1) - i;              // reversed position
    xrhi[r] = h;
    xrlo[r] = (_Float16)(v - (float)h);
  }

#if __has_builtin(__builtin_amdgcn_global_load_async_to_lds_b128)
#if __has_builtin(__builtin_amdgcn_s_wait_asynccnt)
  __builtin_amdgcn_s_wait_asynccnt(0);
#else
  asm volatile("s_wait_asynccnt 0x0" ::: "memory");
#endif
#endif
  __syncthreads();

  const int wave  = tid >> 5;
  const int lane  = tid & 31;
  const int lhalf = lane >> 4;   // 0 | 1
  const int lmod  = lane & 15;

  for (int tile = tile_lo + wave; tile < tile_hi; tile += 8) {
    const int t0 = t_start + (tile << 4);
    v8f c = {};

    for (int jb = 0; jb < KLEN; jb += 32) {
      // ---- A (x frames), f16 16x32 layout. Reversed storage makes each
      //      8-half K-group a contiguous ascending run in LDS.
      v16h ahi, alo;
      const int rbase = (win_len - 1) - (t0 + lmod - win_lo) + jb;  // K=0 rev idx
      #pragma unroll
      for (int e = 0; e < 16; ++e) {
        const int r = rbase + (lhalf << 4) + e;
        ahi[e] = xrhi[r];
        alo[e] = xrlo[r];
      }
      // ---- B (taps), f16 32x16 layout: lane = N, halves = K ascending
      const int boff = lmod * 256 + jb + (lhalf << 4);
      const v16h bhi = *(const v16h*)(khi + boff);
      const v16h blo = *(const v16h*)(klo + boff);

      c = __builtin_amdgcn_wmma_f32_16x16x32_f16(false, ahi, false, bhi, (short)0, c, false, false);
      c = __builtin_amdgcn_wmma_f32_16x16x32_f16(false, ahi, false, blo, (short)0, c, false, false);
      c = __builtin_amdgcn_wmma_f32_16x16x32_f16(false, alo, false, bhi, (short)0, c, false, false);
    }

    // ---- per-row lerp of the two needed frame columns (C: lane=N(+16), vgpr=M%8)
    const int i = lmod;                    // row handled (lanes 16..31 mirror rows)
    const int t = t0 + i;
    const float srcf = ((float)t + 0.5f) * (1.0f / (float)HOP) - 0.5f;
    const float s    = fminf(fmaxf(srcf, 0.0f), (float)(FRAMES - 1));
    const float lof  = floorf(s);
    const int   lo   = (int)lof;
    const float w    = s - lof;
    const int   hi   = min(lo + 1, FRAMES - 1);
    const int   slo  = (lo - f0) + ((i >> 3) << 4);
    const int   shi  = (hi - f0) + ((i >> 3) << 4);
    float ylo = 0.0f, yhi = 0.0f;
    #pragma unroll
    for (int v = 0; v < 8; ++v) {
      const float a0 = __shfl(c[v], slo, 32);
      const float a1 = __shfl(c[v], shi, 32);
      if ((i & 7) == v) { ylo = a0; yhi = a1; }
    }
    if (lane < 16 && t <= t_end) {
      out[(size_t)b * TLEN + t] = (1.0f - w) * ylo + w * yhi;
    }
  }
}

// ---------------------------------------------------------------------------
extern "C" void kernel_launch(void* const* d_in, const int* in_sizes, int n_in,
                              void* d_out, int out_size, void* d_ws, size_t ws_size,
                              hipStream_t stream) {
  (void)in_sizes; (void)n_in; (void)out_size; (void)ws_size;
  const float* ex      = (const float*)d_in[0];   // (4, 96000) f32
  const float* log_mag = (const float*)d_in[1];   // (4, 400, 129) f32
  float*       out     = (float*)d_out;           // (4, 96000) f32

  _Float16* ker_hi = (_Float16*)d_ws;                       // 1600*256 halves
  _Float16* ker_lo = ker_hi + (size_t)BATCH * FRAMES * 256; // 1600*256 halves

  minphase_stage1<<<BATCH * FRAMES, 256, 0, stream>>>(log_mag, ker_hi, ker_lo);
  ltv_fir_stage2<<<dim3(NGRP, BATCH, NSPLIT), 256, 0, stream>>>(ex, ker_hi, ker_lo, out);
}
